// TokAutoDctMultiHeadAttention_5428838662765
// MI455X (gfx1250) — matrix operand order repaired and checked by reading, not compiled, not run on hardware
//
#include <hip/hip_runtime.h>
#include <hip/hip_bf16.h>
#include <math.h>

typedef __attribute__((ext_vector_type(16))) _Float16 v16h;
typedef __attribute__((ext_vector_type(8)))  _Float16 v8h;
typedef __attribute__((ext_vector_type(8)))  float    v8f;

#define B_   2
#define L_   4096
#define DM_  768
#define HD_  128
#define FL_  2049
#define EPS_ 1e-9f
#define INV_SQRT_DK_ 0.08838834764831845f  /* 1/sqrt(128) */

// ---------------------------------------------------------------------------
// Projection GEMM: Y[row, d] = X[row, :768] . W[d, :768] + bias[d]
// One wave per 16x32 output strip (two 16x16 WMMA tiles sharing the A frag).
// ---------------------------------------------------------------------------
__global__ __launch_bounds__(32) void fftattn_proj(const float* __restrict__ X,
                                                   const float* __restrict__ W,
                                                   const float* __restrict__ bias,
                                                   float* __restrict__ Y) {
  const int lane   = threadIdx.x;
  const int rowT   = blockIdx.x;      // over B*L/16 row tiles
  const int colS   = blockIdx.y;      // over 128/32 col strips
  const int n      = lane & 15;
  const int khalf  = lane >> 4;
  const int kbaseA = khalf * 16;
  const int kbaseB = khalf * 16;

  v8f acc0 = {}, acc1 = {};
  const float* Arow = X + (size_t)(rowT * 16 + n) * DM_;
  const float* B0   = W + (size_t)(colS * 32 + n) * DM_;
  const float* B1   = B0 + (size_t)16 * DM_;

  for (int k0 = 0; k0 < DM_; k0 += 32) {
    if (k0 + 32 < DM_) {
      __builtin_prefetch(Arow + k0 + 32, 0, 0);
      __builtin_prefetch(B0 + k0 + 32, 0, 0);
      __builtin_prefetch(B1 + k0 + 32, 0, 0);
    }
    const float4 a0 = *(const float4*)(Arow + k0 + kbaseA);
    const float4 a1 = *(const float4*)(Arow + k0 + kbaseA + 4);
    const float4 a2 = *(const float4*)(Arow + k0 + kbaseA + 8);
    const float4 a3 = *(const float4*)(Arow + k0 + kbaseA + 12);
    v16h a;
    a[0]=(_Float16)a0.x; a[1]=(_Float16)a0.y; a[2]=(_Float16)a0.z; a[3]=(_Float16)a0.w;
    a[4]=(_Float16)a1.x; a[5]=(_Float16)a1.y; a[6]=(_Float16)a1.z; a[7]=(_Float16)a1.w;
    a[8]=(_Float16)a2.x; a[9]=(_Float16)a2.y; a[10]=(_Float16)a2.z; a[11]=(_Float16)a2.w;
    a[12]=(_Float16)a3.x; a[13]=(_Float16)a3.y; a[14]=(_Float16)a3.z; a[15]=(_Float16)a3.w;
    v16h bf0, bf1;
#pragma unroll
    for (int j = 0; j < 4; ++j) {
      const float4 w0 = *(const float4*)(B0 + k0 + kbaseB + 4 * j);
      const float4 w1 = *(const float4*)(B1 + k0 + kbaseB + 4 * j);
      bf0[4*j]=(_Float16)w0.x; bf0[4*j+1]=(_Float16)w0.y;
      bf0[4*j+2]=(_Float16)w0.z; bf0[4*j+3]=(_Float16)w0.w;
      bf1[4*j]=(_Float16)w1.x; bf1[4*j+1]=(_Float16)w1.y;
      bf1[4*j+2]=(_Float16)w1.z; bf1[4*j+3]=(_Float16)w1.w;
    }
    acc0 = __builtin_amdgcn_wmma_f32_16x16x32_f16(false, a, false, bf0,
                                                  (short)0, acc0, false, false);
    acc1 = __builtin_amdgcn_wmma_f32_16x16x32_f16(false, a, false, bf1,
                                                  (short)0, acc1, false, false);
  }

  const int d0 = colS * 32 + n;
  const int d1 = d0 + 16;
  const float bv0 = bias[d0];
  const float bv1 = bias[d1];
#pragma unroll
  for (int r = 0; r < 8; ++r) {
    const int row = rowT * 16 + khalf * 8 + r;
    Y[(size_t)row * HD_ + d0] = acc0[r] + bv0;
    Y[(size_t)row * HD_ + d1] = acc1[r] + bv1;
  }
}

// ---------------------------------------------------------------------------
// Radix-2 Stockham autosort FFT of length 4096 in LDS (ping-pong buffers).
// ---------------------------------------------------------------------------
static __device__ float2* fft4096_lds(float2* s, float2* t, int tid, float sign) {
  int l = 2048, m = 1;
  for (int stage = 0; stage < 12; ++stage) {
    for (int idx = tid; idx < 2048; idx += 256) {
      const int j = idx / m;
      const int k = idx - j * m;
      const float ang = sign * 6.28318530717958647692f * (float)j / (float)(2 * l);
      float sn, cs;
      __sincosf(ang, &sn, &cs);
      const float2 c0 = s[k + j * m];
      const float2 c1 = s[k + j * m + 2048];
      float2 su, di, wd;
      su.x = c0.x + c1.x; su.y = c0.y + c1.y;
      di.x = c0.x - c1.x; di.y = c0.y - c1.y;
      wd.x = cs * di.x - sn * di.y;
      wd.y = cs * di.y + sn * di.x;
      t[k + 2 * j * m]     = su;
      t[k + 2 * j * m + m] = wd;
    }
    __syncthreads();
    float2* tmp = s; s = t; t = tmp;
    l >>= 1; m <<= 1;
  }
  return s;
}

// rFFT: one block per (b, d) channel.
__global__ __launch_bounds__(256) void fftattn_rfft(const float* __restrict__ P,
                                                    float2* __restrict__ T) {
  __shared__ float2 bufA[4096];
  __shared__ float2 bufB[4096];
  const int tid = threadIdx.x;
  const int b   = blockIdx.x / HD_;
  const int d   = blockIdx.x % HD_;
  const float* src = P + (size_t)b * L_ * HD_ + d;
  for (int i = tid; i < L_; i += 256) {
    float2 v; v.x = src[(size_t)i * HD_]; v.y = 0.0f;
    bufA[i] = v;
  }
  __syncthreads();
  float2* res = fft4096_lds(bufA, bufB, tid, -1.0f);
  float2* dst = T + (size_t)(b * HD_ + d) * FL_;
  for (int f = tid; f < FL_; f += 256) dst[f] = res[f];
}

// Selector MLP + Gumbel softmax gate, one block (64 threads) per (b, f) bin.
__global__ __launch_bounds__(64) void fftattn_selector(const float2* __restrict__ T,
                                                       const float* __restrict__ W1,
                                                       const float* __restrict__ b1,
                                                       const float* __restrict__ W2,
                                                       const float* __restrict__ b2,
                                                       const float* __restrict__ noise,
                                                       float* __restrict__ Z) {
  __shared__ float h1[64];
  const int j = threadIdx.x;
  const int b = blockIdx.x / FL_;
  const int f = blockIdx.x % FL_;

  float acc = b1[j];
  const float*  w   = W1 + (size_t)j * HD_;
  const float2* col = T + (size_t)b * HD_ * FL_ + f;
  for (int d = 0; d < HD_; ++d) acc += w[d] * col[(size_t)d * FL_].x;
  h1[j] = fmaxf(acc, 0.0f);
  __syncthreads();

  if (j == 0) {
    float l0 = b2[0], l1 = b2[1];
    for (int t = 0; t < 64; ++t) { l0 += W2[t] * h1[t]; l1 += W2[64 + t] * h1[t]; }
    const float* ns = noise + ((size_t)b * FL_ + f) * 2;
    const float g0 = -__logf(-__logf(ns[0] + EPS_) + EPS_);
    const float g1 = -__logf(-__logf(ns[1] + EPS_) + EPS_);
    const float a0 = l0 + g0, a1 = l1 + g1;        // TEMP == 1
    const float mx = fmaxf(a0, a1);
    const float e0 = __expf(a0 - mx), e1 = __expf(a1 - mx);
    Z[(size_t)b * FL_ + f] = e1 / (e0 + e1);
  }
}

// Gate spectrum, Hermitian-extend, inverse FFT, store transposed f16 [b][l][d].
__global__ __launch_bounds__(256) void fftattn_modirfft(const float2* __restrict__ T,
                                                        const float* __restrict__ Z,
                                                        _Float16* __restrict__ YT) {
  __shared__ float2 bufA[4096];
  __shared__ float2 bufB[4096];
  const int tid = threadIdx.x;
  const int b   = blockIdx.x / HD_;
  const int d   = blockIdx.x % HD_;
  const float2* src = T + (size_t)(b * HD_ + d) * FL_;
  const float*  z   = Z + (size_t)b * FL_;
  for (int f = tid; f < FL_; f += 256) {
    float2 v = src[f];
    const float zz = z[f];
    v.x *= zz; v.y *= zz;
    bufA[f] = v;
    if (f >= 1 && f < FL_ - 1) {              // Hermitian mirror (skip DC, Nyquist)
      float2 c; c.x = v.x; c.y = -v.y;
      bufA[L_ - f] = c;
    }
  }
  __syncthreads();
  float2* res = fft4096_lds(bufA, bufB, tid, 1.0f);
  const float scale = 1.0f / (float)L_;
  for (int l = tid; l < L_; l += 256)
    YT[((size_t)b * L_ + l) * HD_ + d] = (_Float16)(res[l].x * scale);
}

// ---------------------------------------------------------------------------
// Fused attention: per wave, one 16-row tile. Two sweeps over all 64 column
// strips (16x64 WMMA scores per strip): pass 1 accumulates online row max/sum
// (scores stay in registers), pass 2 recomputes and writes the normalized
// probabilities exactly once. Column reduction exploits the WMMA C/D layout:
// VGPR r -> row M = r + 8*(lane>=16), col N = lane&15, so a shfl_xor butterfly
// over each 16-lane half reduces over columns of a single row.
// HBM traffic: one 128MB write + L2-resident operands (qfT/kfT = 2MB each).
// ---------------------------------------------------------------------------
__global__ __launch_bounds__(256) void fftattn_attn(const _Float16* __restrict__ QT,
                                                    const _Float16* __restrict__ KT,
                                                    const int* __restrict__ mask,
                                                    float* __restrict__ out) {
  const int lane   = threadIdx.x & 31;
  const int wid    = threadIdx.x >> 5;
  const int lt     = blockIdx.x * 8 + wid;   // 16-row tile index
  const int b      = blockIdx.y;
  const int n      = lane & 15;
  const int khalf  = lane >> 4;
  const int kbaseA = khalf * 16;
  const int kbaseB = khalf * 16;

  const _Float16* Arow  = QT + ((size_t)b * L_ + lt * 16 + n) * HD_;
  const _Float16* Kbase = KT + (size_t)b * L_ * HD_;
  const int*      mrow  = mask + (size_t)b * L_;
  float* orow = out + ((size_t)b * L_ + lt * 16) * L_;

  // Preload A fragments for all 4 K-steps; reused across all 128 strip sweeps.
  v16h afrag[4];
#pragma unroll
  for (int kk = 0; kk < 4; ++kk) {
    const v8h alo = *(const v8h*)(Arow + kk * 32 + kbaseA);
    const v8h ahi = *(const v8h*)(Arow + kk * 32 + kbaseA + 8);
    afrag[kk] = __builtin_shufflevector(alo, ahi,
        0, 1, 2, 3, 4, 5, 6, 7, 8, 9, 10, 11, 12, 13, 14, 15);
  }

  float runmax[8], runsum[8];
#pragma unroll
  for (int r = 0; r < 8; ++r) { runmax[r] = -3.0e38f; runsum[r] = 0.0f; }

  // ---- pass 1: online max/sum ----
  for (int ms = 0; ms < L_ / 64; ++ms) {
    const _Float16* Brow = Kbase + ((size_t)ms * 64 + n) * HD_;
    v8f acc[4] = {{}, {}, {}, {}};
#pragma unroll
    for (int kk = 0; kk < 4; ++kk) {
#pragma unroll
      for (int t = 0; t < 4; ++t) {
        const v16h bf = *(const v16h*)(Brow + (size_t)t * 16 * HD_ + kk * 32 + kbaseB);
        acc[t] = __builtin_amdgcn_wmma_f32_16x16x32_f16(false, afrag[kk], false, bf,
                                                        (short)0, acc[t], false, false);
      }
    }
    bool dead[4];
#pragma unroll
    for (int t = 0; t < 4; ++t) dead[t] = (mrow[ms * 64 + t * 16 + n] == 0);
#pragma unroll
    for (int r = 0; r < 8; ++r) {
      float vs[4];
      float tm = -3.0e38f;
#pragma unroll
      for (int t = 0; t < 4; ++t) {
        float v = acc[t][r] * INV_SQRT_DK_;
        if (dead[t]) v = -1.0e9f;
        vs[t] = v;
        tm = fmaxf(tm, v);
      }
#pragma unroll
      for (int off = 8; off > 0; off >>= 1)
        tm = fmaxf(tm, __shfl_xor(tm, off, 16));
      const float nm = fmaxf(runmax[r], tm);
      float ts = 0.0f;
#pragma unroll
      for (int t = 0; t < 4; ++t) ts += __expf(vs[t] - nm);
#pragma unroll
      for (int off = 8; off > 0; off >>= 1)
        ts += __shfl_xor(ts, off, 16);
      runsum[r] = runsum[r] * __expf(runmax[r] - nm) + ts;
      runmax[r] = nm;
    }
  }

  float inv[8];
#pragma unroll
  for (int r = 0; r < 8; ++r) inv[r] = 1.0f / runsum[r];

  // ---- pass 2: recompute scores, write normalized probabilities once ----
  for (int ms = 0; ms < L_ / 64; ++ms) {
    const _Float16* Brow = Kbase + ((size_t)ms * 64 + n) * HD_;
    v8f acc[4] = {{}, {}, {}, {}};
#pragma unroll
    for (int kk = 0; kk < 4; ++kk) {
#pragma unroll
      for (int t = 0; t < 4; ++t) {
        const v16h bf = *(const v16h*)(Brow + (size_t)t * 16 * HD_ + kk * 32 + kbaseB);
        acc[t] = __builtin_amdgcn_wmma_f32_16x16x32_f16(false, afrag[kk], false, bf,
                                                        (short)0, acc[t], false, false);
      }
    }
#pragma unroll
    for (int t = 0; t < 4; ++t) {
      const int mcol = ms * 64 + t * 16 + n;
      const bool dead = (mrow[mcol] == 0);
#pragma unroll
      for (int r = 0; r < 8; ++r) {
        float v = acc[t][r] * INV_SQRT_DK_;
        if (dead) v = -1.0e9f;
        orow[(size_t)(khalf * 8 + r) * L_ + mcol] = __expf(v - runmax[r]) * inv[r];
      }
    }
  }
}

// ---------------------------------------------------------------------------
extern "C" void kernel_launch(void* const* d_in, const int* in_sizes, int n_in,
                              void* d_out, int out_size, void* d_ws, size_t ws_size,
                              hipStream_t stream) {
  (void)in_sizes; (void)n_in; (void)out_size; (void)ws_size;

  const float* query  = (const float*)d_in[0];
  const float* key    = (const float*)d_in[1];
  const int*   mask   = (const int*)d_in[2];
  const float* Wq     = (const float*)d_in[3];
  const float* bq     = (const float*)d_in[4];
  const float* Wk     = (const float*)d_in[5];
  const float* bk     = (const float*)d_in[6];
  const float* qW1    = (const float*)d_in[7];
  const float* qb1    = (const float*)d_in[8];
  const float* qW2    = (const float*)d_in[9];
  const float* qb2    = (const float*)d_in[10];
  const float* kW1    = (const float*)d_in[11];
  const float* kb1    = (const float*)d_in[12];
  const float* kW2    = (const float*)d_in[13];
  const float* kb2    = (const float*)d_in[14];
  const float* qnoise = (const float*)d_in[15];
  const float* knoise = (const float*)d_in[16];
  float* out = (float*)d_out;

  char* wsp = (char*)d_ws;
  auto carve = [&](size_t bytes) -> char* {
    char* p = wsp;
    wsp += (bytes + 255) & ~(size_t)255;
    return p;
  };
  float*    qproj = (float*)   carve((size_t)B_ * L_ * HD_ * sizeof(float));
  float*    kproj = (float*)   carve((size_t)B_ * L_ * HD_ * sizeof(float));
  float2*   qtff  = (float2*)  carve((size_t)B_ * HD_ * FL_ * sizeof(float2));
  float2*   ktff  = (float2*)  carve((size_t)B_ * HD_ * FL_ * sizeof(float2));
  float*    zq    = (float*)   carve((size_t)B_ * FL_ * sizeof(float));
  float*    zk    = (float*)   carve((size_t)B_ * FL_ * sizeof(float));
  _Float16* qfT   = (_Float16*)carve((size_t)B_ * L_ * HD_ * sizeof(_Float16));
  _Float16* kfT   = (_Float16*)carve((size_t)B_ * L_ * HD_ * sizeof(_Float16));

  const dim3 gProj(B_ * L_ / 16, HD_ / 32);
  fftattn_proj<<<gProj, 32, 0, stream>>>(query, Wq, bq, qproj);
  fftattn_proj<<<gProj, 32, 0, stream>>>(key,   Wk, bk, kproj);

  fftattn_rfft<<<B_ * HD_, 256, 0, stream>>>(qproj, qtff);
  fftattn_rfft<<<B_ * HD_, 256, 0, stream>>>(kproj, ktff);

  fftattn_selector<<<B_ * FL_, 64, 0, stream>>>(qtff, qW1, qb1, qW2, qb2, qnoise, zq);
  fftattn_selector<<<B_ * FL_, 64, 0, stream>>>(ktff, kW1, kb1, kW2, kb2, knoise, zk);

  fftattn_modirfft<<<B_ * HD_, 256, 0, stream>>>(qtff, zq, qfT);
  fftattn_modirfft<<<B_ * HD_, 256, 0, stream>>>(ktff, zk, kfT);

  const dim3 gAttn(L_ / 16 / 8, B_);
  fftattn_attn<<<gAttn, 256, 0, stream>>>(qfT, kfT, mask, out);
}
